// AMPTransformerClassifier_63608465654501
// MI455X (gfx1250) — matrix operand order repaired and checked by reading, not compiled, not run on hardware
//
#include <hip/hip_runtime.h>

// ---------------------------------------------------------------------------
// CDNA5 (gfx1250) vector types
// ---------------------------------------------------------------------------
typedef __attribute__((ext_vector_type(4)))  _Float16 v4h;
typedef __attribute__((ext_vector_type(8)))  _Float16 v8h;
typedef __attribute__((ext_vector_type(16))) _Float16 v16h;
typedef __attribute__((ext_vector_type(8)))  float    v8f;
typedef __attribute__((ext_vector_type(4)))  unsigned int v4u;
typedef __attribute__((ext_vector_type(8)))  int      v8i;
typedef __attribute__((ext_vector_type(4)))  int      v4i;

#define V16CAT(lo, hi) __builtin_shufflevector((lo), (hi), 0,1,2,3,4,5,6,7,8,9,10,11,12,13,14,15)

// model dims
#define B_    1024
#define L_    100
#define D_    128
#define H_    8
#define NL_   4
#define FF_   512
#define MTOK  (B_ * L_)   // 102400

// Tensor Data Mover availability (6-arg builtin form on this toolchain)
#if defined(__gfx1250__) && __has_builtin(__builtin_amdgcn_tensor_load_to_lds)
#define HAVE_TDM 1
#else
#define HAVE_TDM 0
#endif

// ---------------------------------------------------------------------------
// TDM: load a 64(n) x 32(k) f16 tile from N-major weights (row stride K elems)
// into LDS with 16B padding after each 64B row (LDS row stride = 80B = GLDA*2).
// Descriptor layout per CDNA5 ISA.
// ---------------------------------------------------------------------------
__device__ __forceinline__ void tdm_load_wtile(const _Float16* gsrc, unsigned lds_byte_off, int K) {
#if HAVE_TDM
    unsigned long long ga = (unsigned long long)gsrc;
    v4u g0;
    g0[0] = 1u;                                             // count=1, user D#
    g0[1] = lds_byte_off;                                   // lds_addr
    g0[2] = (unsigned)(ga & 0xFFFFFFFFu);                   // global_addr lo
    g0[3] = (unsigned)((ga >> 32) & 0x01FFFFFFu) | (2u << 30); // global_addr hi | type=2
    v8i g1;
    //      data_size=2B     pad_enable   pad_interval=16DW  pad_amount=4DW
    g1[0] = (int)((1u << 16) | (1u << 20) | (3u << 22) | (3u << 25));
    g1[1] = (int)(((unsigned)K & 0xFFFFu) << 16);           // tensor_dim0[15:0] @ [63:48]
    g1[2] = (int)((((unsigned)K >> 16) & 0xFFFFu) | (64u << 16)); // dim0 hi | tensor_dim1 lo (64 rows)
    g1[3] = (int)(32u << 16);                               // tensor_dim1 hi | tile_dim0=32
    g1[4] = 64;                                             // tile_dim1=64, tile_dim2=0
    g1[5] = (int)(unsigned)K;                               // tensor_dim0_stride lo
    g1[6] = 0;                                              // stride0 hi | stride1 lo
    g1[7] = 0;
    v4i gz4 = {0, 0, 0, 0};
    v8i gz8 = {0, 0, 0, 0, 0, 0, 0, 0};
    __builtin_amdgcn_tensor_load_to_lds(g0, g1, gz4, gz4, gz8, 0);
#else
    (void)gsrc; (void)lds_byte_off; (void)K;
#endif
}

__device__ __forceinline__ unsigned lds_byte_offset(const void* p) {
    // flat shared-aperture addresses carry the LDS offset in the low 32 bits
    return (unsigned)(unsigned long long)p;
}

// ---------------------------------------------------------------------------
// fp32 -> f16 weight conversion with transpose: Wt[n*K + k] = W[k*N + n]
// ---------------------------------------------------------------------------
__global__ void cvtT_kernel(const float* __restrict__ W, _Float16* __restrict__ Wt, int K, int N) {
    int idx = blockIdx.x * 256 + threadIdx.x;
    if (idx >= K * N) return;
    int n = idx / K, k = idx % K;
    Wt[idx] = (_Float16)W[(size_t)k * N + n];
}

// ---------------------------------------------------------------------------
// Embedding + sinusoidal PE + physchem-embedding concat -> E (MTOK x 192) fp32
// ---------------------------------------------------------------------------
__global__ void embed_kernel(const int* __restrict__ tok,
                             const float* __restrict__ tok_emb,
                             const float* __restrict__ phys_emb,
                             float* __restrict__ E) {
    size_t idx = (size_t)blockIdx.x * 256 + threadIdx.x;
    if (idx >= (size_t)MTOK * 192) return;
    int c = (int)(idx % 192);
    size_t row = idx / 192;
    int l = (int)(row % L_);
    int t = tok[row];
    float v;
    if (c < 128) {
        float ang = (float)l * __expf(-0.07195578415606394f * (float)(c & ~1));
        v = tok_emb[t * 128 + c] + ((c & 1) ? __cosf(ang) : __sinf(ang));
    } else {
        v = phys_emb[t * 64 + (c - 128)];
    }
    E[idx] = v;
}

// ---------------------------------------------------------------------------
// GEMM: Y(MxN) = act(X(MxK) @ W(KxN) + bias [+ res]); W given transposed (NxK f16).
// 128x64 workgroup tile, 8 waves, each wave 32x32 (4x v_wmma_f32_16x16x32_f16 / chunk).
// Software pipelined: A double-buffered (reg prefetch), B triple-buffered (TDM one
// chunk ahead), single barrier per K-chunk.
// M%128==0, N%64==0, K%32==0.  act: 0=none 1=relu 2=gelu(exact) 3=tanh
// ---------------------------------------------------------------------------
#define GLDA 40   // padded f16 row stride in LDS

__global__ __launch_bounds__(256)
void gemm_f16wmma_kernel(const float* __restrict__ X, const _Float16* __restrict__ Wt,
                         const float* __restrict__ bias, const float* __restrict__ res,
                         float* __restrict__ Y, int M, int N, int K, int act) {
    __shared__ __align__(16) _Float16 sA[2][128 * GLDA];
    __shared__ __align__(16) _Float16 sB[3][64 * GLDA];

    const int tid  = threadIdx.x;
    const int lane = tid & 31;
    const int wave = tid >> 5;      // 8 waves
    const int wr   = wave >> 1;     // 0..3: 32-row slice
    const int wc   = wave & 1;      // 0..1: 32-col slice
    const int half = lane >> 4;
    const int lm   = lane & 15;
    const int m0   = blockIdx.y * 128;
    const int n0   = blockIdx.x * 64;

    v8f acc00 = {}, acc01 = {}, acc10 = {}, acc11 = {};

    // ---- pipeline prologue: chunk 0 ----
    float4 ra[4];
    #pragma unroll
    for (int i = 0; i < 4; ++i) {
        int idx = i * 256 + tid;
        ra[i] = *(const float4*)(X + (size_t)(m0 + (idx >> 3)) * K + (idx & 7) * 4);
    }
#if HAVE_TDM
    if (wave == 0)
        tdm_load_wtile(Wt + (size_t)n0 * K, lds_byte_offset(sB[0]), K);
#else
    v8h rb = {};
    if (tid < 128)
        rb = *(const v8h*)(Wt + (size_t)(n0 + (tid >> 1)) * K + (tid & 1) * 8);
#endif

    const int NCH = K >> 5;
    for (int c = 0; c < NCH; ++c) {
        const int k0 = c << 5;
        _Float16* A  = sA[c & 1];
        _Float16* Bb = sB[c % 3];

        // commit staged A regs (fp32 -> f16) into LDS
        #pragma unroll
        for (int i = 0; i < 4; ++i) {
            int idx = i * 256 + tid;
            v4h h4;
            h4[0] = (_Float16)ra[i].x; h4[1] = (_Float16)ra[i].y;
            h4[2] = (_Float16)ra[i].z; h4[3] = (_Float16)ra[i].w;
            *(v4h*)(&A[(idx >> 3) * GLDA + (idx & 7) * 4]) = h4;
        }
#if !HAVE_TDM
        if (tid < 128)
            *(v8h*)(&Bb[(tid >> 1) * GLDA + (tid & 1) * 8]) = rb;
#endif
        // issue next chunk's loads (overlaps with this chunk's compute)
        if (c + 1 < NCH) {
            #pragma unroll
            for (int i = 0; i < 4; ++i) {
                int idx = i * 256 + tid;
                ra[i] = *(const float4*)(X + (size_t)(m0 + (idx >> 3)) * K + k0 + 32 + (idx & 7) * 4);
            }
#if HAVE_TDM
            if (wave == 0)
                tdm_load_wtile(Wt + (size_t)n0 * K + k0 + 32,
                               lds_byte_offset(sB[(c + 1) % 3]), K);
#else
            if (tid < 128)
                rb = *(const v8h*)(Wt + (size_t)(n0 + (tid >> 1)) * K + k0 + 32 + (tid & 1) * 8);
            __builtin_prefetch(X + (size_t)(m0 + (tid >> 1)) * K + k0 + 32, 0, 1);
#endif
        }
#if HAVE_TDM
        if (wave == 0) {
            if (c + 1 < NCH) __builtin_amdgcn_s_wait_tensorcnt(1);  // current buf done
            else             __builtin_amdgcn_s_wait_tensorcnt(0);
        }
#endif
        __syncthreads();

        // fragments + 4 WMMAs
        v8h a0lo = *(const v8h*)(&A[(wr * 32      + lm) * GLDA + half * 8]);
        v8h a0hi = *(const v8h*)(&A[(wr * 32      + lm) * GLDA + half * 8 + 16]);
        v8h a1lo = *(const v8h*)(&A[(wr * 32 + 16 + lm) * GLDA + half * 8]);
        v8h a1hi = *(const v8h*)(&A[(wr * 32 + 16 + lm) * GLDA + half * 8 + 16]);
        v16h a0 = V16CAT(a0lo, a0hi);
        v16h a1 = V16CAT(a1lo, a1hi);
        v8h b0lo = *(const v8h*)(&Bb[(wc * 32      + lm) * GLDA + half * 8]);
        v8h b0hi = *(const v8h*)(&Bb[(wc * 32      + lm) * GLDA + half * 8 + 16]);
        v8h b1lo = *(const v8h*)(&Bb[(wc * 32 + 16 + lm) * GLDA + half * 8]);
        v8h b1hi = *(const v8h*)(&Bb[(wc * 32 + 16 + lm) * GLDA + half * 8 + 16]);
        v16h b0 = V16CAT(b0lo, b0hi);
        v16h b1 = V16CAT(b1lo, b1hi);

        acc00 = __builtin_amdgcn_wmma_f32_16x16x32_f16(false, a0, false, b0, (short)0, acc00, false, false);
        acc01 = __builtin_amdgcn_wmma_f32_16x16x32_f16(false, a0, false, b1, (short)0, acc01, false, false);
        acc10 = __builtin_amdgcn_wmma_f32_16x16x32_f16(false, a1, false, b0, (short)0, acc10, false, false);
        acc11 = __builtin_amdgcn_wmma_f32_16x16x32_f16(false, a1, false, b1, (short)0, acc11, false, false);
        // no closing barrier: A double-buffer + B triple-buffer make next-iteration
        // stores / TDM target buffers whose readers finished before the last barrier.
    }

    // ---- epilogue: bias + residual + activation ----
    #pragma unroll
    for (int ri = 0; ri < 2; ++ri) {
        #pragma unroll
        for (int ti = 0; ti < 2; ++ti) {
            v8f acc = ri ? (ti ? acc11 : acc10) : (ti ? acc01 : acc00);
            int col = n0 + wc * 32 + ti * 16 + lm;
            float bv = bias ? bias[col] : 0.0f;
            #pragma unroll
            for (int r = 0; r < 8; ++r) {
                int row = m0 + wr * 32 + ri * 16 + half * 8 + r;
                float v = acc[r] + bv;
                if (res) v += res[(size_t)row * N + col];
                if (act == 1)      v = fmaxf(v, 0.0f);
                else if (act == 2) v = 0.5f * v * (1.0f + erff(v * 0.70710678118f));
                else if (act == 3) v = tanhf(v);
                Y[(size_t)row * N + col] = v;
            }
        }
    }
}

// ---------------------------------------------------------------------------
// Attention: one 2-wave block per (b,h).  QKV (MTOK x 384) fp32 -> OUT (MTOK x 128)
// ---------------------------------------------------------------------------
#define LP 112   // L padded to 7 tiles of 16

__global__ __launch_bounds__(64)
void attention_kernel(const float* __restrict__ QKV, const int* __restrict__ mask,
                      float* __restrict__ OUT) {
    const int bh   = blockIdx.x;
    const int b    = bh >> 3;       // H_ = 8
    const int h    = bh & 7;
    const int tid  = threadIdx.x;   // 0..63
    const int lane = tid & 31;
    const int wv   = tid >> 5;      // 0..1
    const int half = lane >> 4;
    const int lm   = lane & 15;

    __shared__ __align__(16) _Float16 sQ[LP * 40];
    __shared__ __align__(16) _Float16 sK[LP * 40];
    __shared__ __align__(16) _Float16 sVT[16 * 136];
    __shared__ __align__(16) float    sS[2][16 * 113];
    __shared__ __align__(16) _Float16 sP[2][16 * 136];
    __shared__ float sMB[LP];

    const size_t rowbase = (size_t)b * L_;

    for (int i = tid; i < LP * 32; i += 64) {
        int l = i >> 5, d = i & 31;
        float qv = 0.0f, kv = 0.0f;
        if (d < 16 && l < L_) {
            const float* p = QKV + (rowbase + l) * 384 + h * 16 + d;
            qv = p[0];
            kv = p[128];
        }
        sQ[l * 40 + d] = (_Float16)qv;
        sK[l * 40 + d] = (_Float16)kv;
    }
    for (int i = tid; i < 16 * 128; i += 64) {
        int d = i >> 7, m = i & 127;
        float vv = (m < L_) ? QKV[(rowbase + m) * 384 + 256 + h * 16 + d] : 0.0f;
        sVT[d * 136 + m] = (_Float16)vv;
    }
    for (int i = tid; i < 2 * 16 * 136; i += 64) ((_Float16*)sP)[i] = (_Float16)0.0f;
    for (int i = tid; i < LP; i += 64)
        sMB[i] = (i < L_ && mask[rowbase + i] != 0) ? 0.0f : -3.0e38f;
    __syncthreads();

    for (int it = 0; it < 4; ++it) {
        int tile = it * 2 + wv;
        bool active = (tile < 7);
        if (active) {
            v8h alo = *(const v8h*)(&sQ[(tile * 16 + lm) * 40 + half * 8]);
            v8h ahi = *(const v8h*)(&sQ[(tile * 16 + lm) * 40 + half * 8 + 16]);
            v16h a = V16CAT(alo, ahi);
            #pragma unroll
            for (int jt = 0; jt < 7; ++jt) {
                v8h blo = *(const v8h*)(&sK[(jt * 16 + lm) * 40 + half * 8]);
                v8h bhi = *(const v8h*)(&sK[(jt * 16 + lm) * 40 + half * 8 + 16]);
                v16h bmat = V16CAT(blo, bhi);
                v8f s = {};
                s = __builtin_amdgcn_wmma_f32_16x16x32_f16(false, a, false, bmat, (short)0, s, false, false);
                #pragma unroll
                for (int r = 0; r < 8; ++r)
                    sS[wv][(half * 8 + r) * 113 + jt * 16 + lm] = s[r] * 0.25f;  // HD^-0.5
            }
        }
        __syncthreads();
        if (active && lane < 16) {
            int row = lane;
            float mx = -3.0e38f;
            for (int m = 0; m < LP; ++m) {
                float sv = sS[wv][row * 113 + m] + sMB[m];
                mx = fmaxf(mx, sv);
            }
            float sum = 0.0f;
            for (int m = 0; m < LP; ++m) {
                float sv = sS[wv][row * 113 + m] + sMB[m];
                float p = __expf(sv - mx);
                sS[wv][row * 113 + m] = p;
                sum += p;
            }
            float inv = 1.0f / sum;
            for (int m = 0; m < LP; ++m)
                sP[wv][row * 136 + m] = (_Float16)(sS[wv][row * 113 + m] * inv);
        }
        __syncthreads();
        if (active) {
            v8f o = {};
            #pragma unroll
            for (int kc = 0; kc < 128; kc += 32) {
                v8h alo = *(const v8h*)(&sP[wv][lm * 136 + kc + half * 8]);
                v8h ahi = *(const v8h*)(&sP[wv][lm * 136 + kc + half * 8 + 16]);
                v16h a = V16CAT(alo, ahi);
                v8h blo = *(const v8h*)(&sVT[lm * 136 + kc + half * 8]);
                v8h bhi = *(const v8h*)(&sVT[lm * 136 + kc + half * 8 + 16]);
                v16h bmat = V16CAT(blo, bhi);
                o = __builtin_amdgcn_wmma_f32_16x16x32_f16(false, a, false, bmat, (short)0, o, false, false);
            }
            #pragma unroll
            for (int r = 0; r < 8; ++r) {
                int l = tile * 16 + half * 8 + r;
                if (l < L_) OUT[(rowbase + l) * 128 + h * 16 + lm] = o[r];
            }
        }
        __syncthreads();
    }
}

// ---------------------------------------------------------------------------
// LayerNorm over rows of 128, one wave per row
// ---------------------------------------------------------------------------
__global__ __launch_bounds__(256)
void layernorm_kernel(const float* __restrict__ Yin, const float* __restrict__ g,
                      const float* __restrict__ bta, float* __restrict__ Xout, int rows) {
    int wave = threadIdx.x >> 5, lane = threadIdx.x & 31;
    int row = blockIdx.x * 8 + wave;
    if (row >= rows) return;
    float4 v = *(const float4*)(Yin + (size_t)row * 128 + lane * 4);
    float s = v.x + v.y + v.z + v.w;
    float q = v.x * v.x + v.y * v.y + v.z * v.z + v.w * v.w;
    #pragma unroll
    for (int off = 16; off > 0; off >>= 1) {
        s += __shfl_xor(s, off, 32);
        q += __shfl_xor(q, off, 32);
    }
    float mean = s * (1.0f / 128.0f);
    float var  = q * (1.0f / 128.0f) - mean * mean;
    float rstd = rsqrtf(var + 1e-5f);
    float4 gg = *(const float4*)(g + lane * 4);
    float4 bb = *(const float4*)(bta + lane * 4);
    float4 o;
    o.x = (v.x - mean) * rstd * gg.x + bb.x;
    o.y = (v.y - mean) * rstd * gg.y + bb.y;
    o.z = (v.z - mean) * rstd * gg.z + bb.z;
    o.w = (v.w - mean) * rstd * gg.w + bb.w;
    *(float4*)(Xout + (size_t)row * 128 + lane * 4) = o;
}

// ---------------------------------------------------------------------------
__global__ __launch_bounds__(256)
void rowdot256_kernel(const float* __restrict__ T, const float* __restrict__ w,
                      const float* __restrict__ b2, float* __restrict__ s, int rows) {
    int wave = threadIdx.x >> 5, lane = threadIdx.x & 31;
    int row = blockIdx.x * 8 + wave;
    if (row >= rows) return;
    const float* p = T + (size_t)row * 256;
    float acc = 0.0f;
    #pragma unroll
    for (int i = 0; i < 8; ++i) acc += p[lane + i * 32] * w[lane + i * 32];
    #pragma unroll
    for (int off = 16; off > 0; off >>= 1) acc += __shfl_xor(acc, off, 32);
    if (lane == 0) s[row] = acc + b2[0];
}

// ---------------------------------------------------------------------------
__global__ __launch_bounds__(128)
void pool_kernel(const float* __restrict__ s, const int* __restrict__ mask,
                 const float* __restrict__ X, float* __restrict__ pooled) {
    int b = blockIdx.x, t = threadIdx.x;
    __shared__ float sw[128];
    __shared__ float red[128];
    float sv = -3.0e38f;
    if (t < L_ && mask[b * L_ + t] != 0) sv = s[b * L_ + t];
    sw[t] = sv; red[t] = sv;
    __syncthreads();
    for (int off = 64; off > 0; off >>= 1) {
        if (t < off) red[t] = fmaxf(red[t], red[t + off]);
        __syncthreads();
    }
    float mx = red[0];
    __syncthreads();
    float p = (sw[t] > -1.0e38f) ? __expf(sw[t] - mx) : 0.0f;
    sw[t] = p; red[t] = p;
    __syncthreads();
    for (int off = 64; off > 0; off >>= 1) {
        if (t < off) red[t] += red[t + off];
        __syncthreads();
    }
    float inv = 1.0f / red[0];
    float acc = 0.0f;
    for (int l = 0; l < L_; ++l)
        acc += sw[l] * X[((size_t)b * L_ + l) * 128 + t];
    pooled[b * 128 + t] = acc * inv;
}

// ---------------------------------------------------------------------------
__global__ void pc1_kernel(const float* __restrict__ phys, const float* __restrict__ w,
                           const float* __restrict__ bias, float* __restrict__ out) {
    int idx = blockIdx.x * 256 + threadIdx.x;
    if (idx >= B_ * 128) return;
    int b = idx >> 7, j = idx & 127;
    float acc = bias[j];
    #pragma unroll
    for (int i = 0; i < 8; ++i) acc += phys[b * 8 + i] * w[i * 128 + j];
    out[idx] = fmaxf(acc, 0.0f);
}

__global__ void concat_kernel(const float* __restrict__ a, const float* __restrict__ c,
                              float* __restrict__ f) {
    int idx = blockIdx.x * 256 + threadIdx.x;
    if (idx >= B_ * 384) return;
    int b = idx / 384, col = idx % 384;
    f[idx] = (col < 128) ? a[b * 128 + col] : c[b * 256 + (col - 128)];
}

__global__ void bn_kernel(float* __restrict__ h, const float* __restrict__ g,
                          const float* __restrict__ bb, const float* __restrict__ m,
                          const float* __restrict__ v, int n, int C) {
    int idx = blockIdx.x * 256 + threadIdx.x;
    if (idx >= n) return;
    int c = idx % C;
    h[idx] = (h[idx] - m[c]) * rsqrtf(v[c] + 1e-5f) * g[c] + bb[c];
}

__global__ __launch_bounds__(256)
void final_kernel(const float* __restrict__ h, const float* __restrict__ w,
                  const float* __restrict__ b, float* __restrict__ out) {
    int wave = threadIdx.x >> 5, lane = threadIdx.x & 31;
    int row = blockIdx.x * 8 + wave;
    if (row >= B_) return;
    float acc = 0.0f;
    #pragma unroll
    for (int i = 0; i < 4; ++i) acc += h[(size_t)row * 128 + lane + i * 32] * w[lane + i * 32];
    #pragma unroll
    for (int off = 16; off > 0; off >>= 1) acc += __shfl_xor(acc, off, 32);
    if (lane == 0) out[row] = acc + b[0];
}

// ---------------------------------------------------------------------------
// host orchestration
// ---------------------------------------------------------------------------
static inline void launch_gemm(const float* X, const _Float16* Wt, const float* bias,
                               const float* res, float* Y, int M, int N, int K, int act,
                               hipStream_t s) {
    dim3 g(N / 64, M / 128);
    gemm_f16wmma_kernel<<<g, dim3(256), 0, s>>>(X, Wt, bias, res, Y, M, N, K, act);
}
static inline void launch_cvtT(const float* W, _Float16* Wt, int K, int N, hipStream_t st) {
    cvtT_kernel<<<(K * N + 255) / 256, 256, 0, st>>>(W, Wt, K, N);
}

extern "C" void kernel_launch(void* const* d_in, const int* in_sizes, int n_in,
                              void* d_out, int out_size, void* d_ws, size_t ws_size,
                              hipStream_t stream) {
    (void)in_sizes; (void)n_in; (void)out_size; (void)ws_size;

    const int*   seq      = (const int*)  d_in[0];
    const float* phys     = (const float*)d_in[1];
    const int*   mask     = (const int*)  d_in[2];
    const float* tok_emb  = (const float*)d_in[3];
    const float* phys_emb = (const float*)d_in[4];
    const float* fuse_w   = (const float*)d_in[5];
    const float* fuse_b   = (const float*)d_in[6];
    const float* eln_g    = (const float*)d_in[7];
    const float* eln_b    = (const float*)d_in[8];
    const float* qkv_w    = (const float*)d_in[9];
    const float* qkv_b    = (const float*)d_in[10];
    const float* proj_w   = (const float*)d_in[11];
    const float* proj_b   = (const float*)d_in[12];
    const float* n1_g     = (const float*)d_in[13];
    const float* n1_b     = (const float*)d_in[14];
    const float* ffn_w1   = (const float*)d_in[15];
    const float* ffn_b1   = (const float*)d_in[16];
    const float* ffn_w2   = (const float*)d_in[17];
    const float* ffn_b2   = (const float*)d_in[18];
    const float* n2_g     = (const float*)d_in[19];
    const float* n2_b     = (const float*)d_in[20];
    const float* pool_w1  = (const float*)d_in[21];
    const float* pool_b1  = (const float*)d_in[22];
    const float* pool_w2  = (const float*)d_in[23];
    const float* pool_b2  = (const float*)d_in[24];
    const float* pc_w1    = (const float*)d_in[25];
    const float* pc_b1    = (const float*)d_in[26];
    const float* pc_w2    = (const float*)d_in[27];
    const float* pc_b2    = (const float*)d_in[28];
    const float* cls_w1   = (const float*)d_in[29];
    const float* cls_b1   = (const float*)d_in[30];
    const float* bn1_g    = (const float*)d_in[31];
    const float* bn1_b    = (const float*)d_in[32];
    const float* bn1_m    = (const float*)d_in[33];
    const float* bn1_v    = (const float*)d_in[34];
    const float* cls_w2   = (const float*)d_in[35];
    const float* cls_b2   = (const float*)d_in[36];
    const float* bn2_g    = (const float*)d_in[37];
    const float* bn2_b    = (const float*)d_in[38];
    const float* bn2_m    = (const float*)d_in[39];
    const float* bn2_v    = (const float*)d_in[40];
    const float* cls_w3   = (const float*)d_in[41];
    const float* cls_b3   = (const float*)d_in[42];
    const float* cls_w4   = (const float*)d_in[43];
    const float* cls_b4   = (const float*)d_in[44];

    // ---- workspace layout ----
    char* ws = (char*)d_ws;
    _Float16* wf = (_Float16*)ws;           // f16 transposed-weight arena (elem offsets)
    const size_t FW_FUSE = 0;               // 192*128
    const size_t FW_QKV  = 24576;           // 4*128*384
    const size_t FW_PROJ = 221184;          // 4*128*128
    const size_t FW_F1   = 286720;          // 4*128*512
    const size_t FW_F2   = 548864;          // 4*512*128
    const size_t FW_P1   = 811008;          // 128*256
    const size_t FW_PC2  = 843776;          // 128*256
    const size_t FW_C1   = 876544;          // 384*512
    const size_t FW_C2   = 1073152;         // 512*256
    const size_t FW_C3   = 1204224;         // 256*128

    const size_t OFF_X = 4ull << 20;
    const size_t OFF_B = OFF_X + 52428800ull;
    const size_t OFF_C = OFF_B + 209715200ull;
    const size_t OFF_D = OFF_C + 78643200ull;
    const size_t OFF_S = OFF_D + 52428800ull;
    float* X  = (float*)(ws + OFF_X);
    float* RB = (float*)(ws + OFF_B);
    float* RC = (float*)(ws + OFF_C);
    float* RD = (float*)(ws + OFF_D);
    float* Sv     = (float*)(ws + OFF_S + 0);
    float* Pooled = (float*)(ws + OFF_S + 524288);
    float* PC1    = (float*)(ws + OFF_S + 1048576);
    float* PC2    = (float*)(ws + OFF_S + 1572864);
    float* FEAT   = (float*)(ws + OFF_S + 2621440);
    float* H1     = (float*)(ws + OFF_S + 4194304);
    float* H2     = (float*)(ws + OFF_S + 6291456);
    float* H3     = (float*)(ws + OFF_S + 7340032);

    // ---- weight conversion (transposed to N-major f16) ----
    launch_cvtT(fuse_w, wf + FW_FUSE, 192, 128, stream);
    for (int l = 0; l < NL_; ++l) {
        launch_cvtT(qkv_w  + (size_t)l * 128 * 384, wf + FW_QKV  + (size_t)l * 128 * 384, 128, 384, stream);
        launch_cvtT(proj_w + (size_t)l * 128 * 128, wf + FW_PROJ + (size_t)l * 128 * 128, 128, 128, stream);
        launch_cvtT(ffn_w1 + (size_t)l * 128 * 512, wf + FW_F1   + (size_t)l * 128 * 512, 128, 512, stream);
        launch_cvtT(ffn_w2 + (size_t)l * 512 * 128, wf + FW_F2   + (size_t)l * 512 * 128, 512, 128, stream);
    }
    launch_cvtT(pool_w1, wf + FW_P1,  128, 256, stream);
    launch_cvtT(pc_w2,   wf + FW_PC2, 128, 256, stream);
    launch_cvtT(cls_w1,  wf + FW_C1,  384, 512, stream);
    launch_cvtT(cls_w2,  wf + FW_C2,  512, 256, stream);
    launch_cvtT(cls_w3,  wf + FW_C3,  256, 128, stream);

    // ---- embedding + fuse + encoder layernorm ----
    embed_kernel<<<(MTOK * 192 + 255) / 256, 256, 0, stream>>>(seq, tok_emb, phys_emb, RC);
    launch_gemm(RC, wf + FW_FUSE, fuse_b, nullptr, RD, MTOK, 128, 192, 0, stream);
    layernorm_kernel<<<MTOK / 8, 256, 0, stream>>>(RD, eln_g, eln_b, X, MTOK);

    // ---- transformer layers ----
    for (int l = 0; l < NL_; ++l) {
        launch_gemm(X, wf + FW_QKV + (size_t)l * 128 * 384, qkv_b + l * 384, nullptr,
                    RB, MTOK, 384, 128, 0, stream);
        attention_kernel<<<B_ * H_, 64, 0, stream>>>(RB, mask, RD);
        launch_gemm(RD, wf + FW_PROJ + (size_t)l * 128 * 128, proj_b + l * 128, X,
                    RC, MTOK, 128, 128, 0, stream);
        layernorm_kernel<<<MTOK / 8, 256, 0, stream>>>(RC, n1_g + l * 128, n1_b + l * 128, X, MTOK);
        launch_gemm(X, wf + FW_F1 + (size_t)l * 128 * 512, ffn_b1 + l * 512, nullptr,
                    RB, MTOK, 512, 128, 2 /*gelu*/, stream);
        launch_gemm(RB, wf + FW_F2 + (size_t)l * 512 * 128, ffn_b2 + l * 128, X,
                    RC, MTOK, 128, 512, 0, stream);
        layernorm_kernel<<<MTOK / 8, 256, 0, stream>>>(RC, n2_g + l * 128, n2_b + l * 128, X, MTOK);
    }

    // ---- attention pooling ----
    launch_gemm(X, wf + FW_P1, pool_b1, nullptr, RB, MTOK, 256, 128, 3 /*tanh*/, stream);
    rowdot256_kernel<<<MTOK / 8, 256, 0, stream>>>(RB, pool_w2, pool_b2, Sv, MTOK);
    pool_kernel<<<B_, 128, 0, stream>>>(Sv, mask, X, Pooled);

    // ---- physchem branch ----
    pc1_kernel<<<(B_ * 128 + 255) / 256, 256, 0, stream>>>(phys, pc_w1, pc_b1, PC1);
    launch_gemm(PC1, wf + FW_PC2, pc_b2, nullptr, PC2, B_, 256, 128, 1 /*relu*/, stream);

    // ---- classifier head ----
    concat_kernel<<<(B_ * 384 + 255) / 256, 256, 0, stream>>>(Pooled, PC2, FEAT);
    launch_gemm(FEAT, wf + FW_C1, cls_b1, nullptr, H1, B_, 512, 384, 1, stream);
    bn_kernel<<<(B_ * 512 + 255) / 256, 256, 0, stream>>>(H1, bn1_g, bn1_b, bn1_m, bn1_v, B_ * 512, 512);
    launch_gemm(H1, wf + FW_C2, cls_b2, nullptr, H2, B_, 256, 512, 1, stream);
    bn_kernel<<<(B_ * 256 + 255) / 256, 256, 0, stream>>>(H2, bn2_g, bn2_b, bn2_m, bn2_v, B_ * 256, 256);
    launch_gemm(H2, wf + FW_C3, cls_b3, nullptr, H3, B_, 128, 256, 1, stream);
    final_kernel<<<B_ / 8, 256, 0, stream>>>(H3, cls_w4, cls_b4, (float*)d_out);
}
